// Attention_35021163332119
// MI455X (gfx1250) — matrix operand order repaired and checked by reading, not compiled, not run on hardware
//
#include <hip/hip_runtime.h>
#include <hip/hip_bf16.h>

// ---------------------------------------------------------------------------
// Problem constants (match reference): B=2, N=2048, DIM=1024, H=16, DH=64
// ---------------------------------------------------------------------------
constexpr int Bc   = 2;
constexpr int Nc   = 2048;
constexpr int DIMc = 1024;
constexpr int Hc   = 16;
constexpr int DHc  = 64;

typedef __attribute__((ext_vector_type(16))) __bf16 v16bf;
typedef __attribute__((ext_vector_type(8)))  __bf16 v8bf;
typedef __attribute__((ext_vector_type(8)))  float  v8f;

#define WMMA_BF16(a, b, c) \
  __builtin_amdgcn_wmma_f32_16x16x32_bf16(false, (a), false, (b), (short)0, (c), false, false)

__device__ __forceinline__ __bf16 f2bf(float f) { return (__bf16)f; }

// Concatenate two 8-wide bf16 vectors -> 16-wide (register placement only).
__device__ __forceinline__ v16bf cat16(v8bf lo, v8bf hi) {
  return __builtin_shufflevector(lo, hi, 0, 1, 2, 3, 4, 5, 6, 7,
                                 8, 9, 10, 11, 12, 13, 14, 15);
}
// 32 contiguous bytes -> v16bf (two b128 loads).
__device__ __forceinline__ v16bf ld16c(const __bf16* p) {
  return cat16(*(const v8bf*)p, *(const v8bf*)(p + 8));
}

__device__ __forceinline__ void async_copy16(unsigned lds_addr, const void* gsrc) {
  asm volatile("global_load_async_to_lds_b128 %0, %1, off"
               :: "v"(lds_addr), "v"((unsigned long long)(size_t)gsrc)
               : "memory");
}
__device__ __forceinline__ void wait_async0() {
  asm volatile("s_wait_asynccnt 0" ::: "memory");
}

__device__ __forceinline__ float hmax16(float v) {
  v = fmaxf(v, __shfl_xor(v, 1, 16));
  v = fmaxf(v, __shfl_xor(v, 2, 16));
  v = fmaxf(v, __shfl_xor(v, 4, 16));
  v = fmaxf(v, __shfl_xor(v, 8, 16));
  return v;
}
__device__ __forceinline__ float hsum16(float v) {
  v += __shfl_xor(v, 1, 16);
  v += __shfl_xor(v, 2, 16);
  v += __shfl_xor(v, 4, 16);
  v += __shfl_xor(v, 8, 16);
  return v;
}

// ---------------------------------------------------------------------------
// 1) f32 -> bf16 cast
// ---------------------------------------------------------------------------
__global__ void cast_f32_bf16_kernel(const float* __restrict__ in,
                                     __bf16* __restrict__ out, int n) {
  int i = blockIdx.x * blockDim.x + threadIdx.x;
  if (i < n) out[i] = f2bf(in[i]);
}

// ---------------------------------------------------------------------------
// 2) Double-buffered LDS-tiled bf16 WMMA GEMM: C = A*B (+bias), f32 out.
//    Block tile 64(M) x 128(N), K stepped by 32. 8 waves as 2(M) x 4(N),
//    each wave computes a 32x32 C tile (4 WMMAs/step, every fragment
//    feeds two WMMAs). A tile k+1 is copied global->LDS with the CDNA5
//    async-to-LDS path while tile k's WMMAs run; B tile k+1 is loaded to
//    registers in the shadow of the WMMAs and scattered transposed into
//    the next LDS buffer. One barrier per K-step.
//    A-fragment k order used here (16-bit A 16x32, B the same):
//      lane<16 : K = 0..15 ; lane>=16 : K = 16..31
//    C layout: vgpr j -> M = j + 8*(lane/16), N = lane%16
// ---------------------------------------------------------------------------
__global__ void __launch_bounds__(256)
gemm_bf16_wmma(const __bf16* __restrict__ A, const __bf16* __restrict__ Bm,
               const float* __restrict__ bias, float* __restrict__ C,
               int M, int K, int Nn) {
  __shared__ __bf16 As[2][64 * 32];   // [m][k]
  __shared__ __bf16 Bt[2][128 * 32];  // [n][k] (transposed tile)

  const int lane = threadIdx.x & 31;
  const int wave = threadIdx.x >> 5;
  const int half = lane >> 4;
  const int l16  = lane & 15;
  const int t    = threadIdx.x;

  const int bm = blockIdx.y * 64;
  const int bn = blockIdx.x * 128;
  const int wm = (wave >> 2) * 32;  // 0 or 32
  const int wn = (wave & 3) * 32;   // 0,32,64,96

  // Cooperative loader indices:
  const int ar = t >> 2, ak = (t & 3) * 8;   // A: row 0..63, k-chunk of 8
  const int bk = t >> 4, bc = (t & 15) * 8;  // B: k-row 0..15 (x2), col-chunk of 8

  const __bf16* aSrc = A + (size_t)(bm + ar) * K + ak;
  const unsigned aDst[2] = {(unsigned)(size_t)&As[0][ar * 32 + ak],
                            (unsigned)(size_t)&As[1][ar * 32 + ak]};

  v8f c00 = {}, c01 = {}, c10 = {}, c11 = {};

  // ---- prologue: fill buffer 0 ----
  {
    async_copy16(aDst[0], aSrc);
    v8bf r0 = *(const v8bf*)(Bm + (size_t)bk * Nn + bn + bc);
    v8bf r1 = *(const v8bf*)(Bm + (size_t)(16 + bk) * Nn + bn + bc);
    __bf16* bt = &Bt[0][0];
#pragma unroll
    for (int e = 0; e < 8; ++e) bt[(bc + e) * 32 + bk] = r0[e];
#pragma unroll
    for (int e = 0; e < 8; ++e) bt[(bc + e) * 32 + 16 + bk] = r1[e];
  }

  const int nk = K >> 5;
  for (int kt = 0; kt < nk; ++kt) {
    wait_async0();       // own async A copy into buf[cur] done
    __syncthreads();     // all waves' Bt/As writes visible; prior reads done
    const int cur = kt & 1, nxt = cur ^ 1;
    const bool more = (kt + 1) < nk;

    v8bf r0 = {}, r1 = {};
    if (more) {
      const int k1 = (kt + 1) * 32;
      async_copy16(aDst[nxt], aSrc + k1);
      r0 = *(const v8bf*)(Bm + (size_t)(k1 + bk) * Nn + bn + bc);
      r1 = *(const v8bf*)(Bm + (size_t)(k1 + 16 + bk) * Nn + bn + bc);
      __builtin_prefetch(Bm + (size_t)(k1 + 32 + bk) * Nn + bn + bc, 0, 1);
    }

    const __bf16* a0p = &As[cur][(wm + l16) * 32];
    const __bf16* a1p = &As[cur][(wm + 16 + l16) * 32];
    v16bf a0 = cat16(*(const v8bf*)(a0p + half * 16),
                     *(const v8bf*)(a0p + half * 16 + 8));
    v16bf a1 = cat16(*(const v8bf*)(a1p + half * 16),
                     *(const v8bf*)(a1p + half * 16 + 8));
    v16bf b0 = ld16c(&Bt[cur][(wn + l16) * 32 + half * 16]);
    v16bf b1 = ld16c(&Bt[cur][(wn + 16 + l16) * 32 + half * 16]);
    c00 = WMMA_BF16(a0, b0, c00);
    c01 = WMMA_BF16(a0, b1, c01);
    c10 = WMMA_BF16(a1, b0, c10);
    c11 = WMMA_BF16(a1, b1, c11);

    if (more) {
      __bf16* bt = &Bt[nxt][0];
#pragma unroll
      for (int e = 0; e < 8; ++e) bt[(bc + e) * 32 + bk] = r0[e];
#pragma unroll
      for (int e = 0; e < 8; ++e) bt[(bc + e) * 32 + 16 + bk] = r1[e];
    }
  }

  // ---- epilogue ----
  const float bv0 = bias ? bias[bn + wn + l16] : 0.0f;
  const float bv1 = bias ? bias[bn + wn + 16 + l16] : 0.0f;
#pragma unroll
  for (int j = 0; j < 8; ++j) {
    const int m0 = bm + wm + j + 8 * half;
    const int m1 = m0 + 16;
    C[(size_t)m0 * Nn + bn + wn + l16]      = c00[j] + bv0;
    C[(size_t)m0 * Nn + bn + wn + 16 + l16] = c01[j] + bv1;
    C[(size_t)m1 * Nn + bn + wn + l16]      = c10[j] + bv0;
    C[(size_t)m1 * Nn + bn + wn + 16 + l16] = c11[j] + bv1;
  }
}

// ---------------------------------------------------------------------------
// 3) RoPE + split into head-major bf16 Q/K/V  [B*H][N][DH]
// ---------------------------------------------------------------------------
__global__ void rope_split_kernel(const float* __restrict__ qkv,
                                  __bf16* __restrict__ qh,
                                  __bf16* __restrict__ kh,
                                  __bf16* __restrict__ vh) {
  const int total = Bc * Nc * Hc * (DHc / 2);
  int idx = blockIdx.x * blockDim.x + threadIdx.x;
  if (idx >= total) return;
  const int d2 = idx & 31;  int t = idx >> 5;
  const int h  = t & 15;        t >>= 4;
  const int n  = t & 2047;
  const int b  = t >> 11;

  const size_t base = ((size_t)b * Nc + n) * (3 * DIMc) + h * DHc + 2 * d2;
  const float q0 = qkv[base],            q1 = qkv[base + 1];
  const float k0 = qkv[base + DIMc],     k1 = qkv[base + DIMc + 1];
  const float v0 = qkv[base + 2 * DIMc], v1 = qkv[base + 2 * DIMc + 1];

  const float inv = __expf(-(float)(2 * d2) * (1.0f / 64.0f) * 9.210340371976184f);
  const float fr = (float)n * inv;
  const float c = __cosf(fr), s = __sinf(fr);
  const float scale = 0.125f;  // DH^-0.5 folded into Q

  const size_t ob = ((size_t)(b * Hc + h) * Nc + n) * DHc + 2 * d2;
  qh[ob]     = f2bf((q0 * c - q1 * s) * scale);
  qh[ob + 1] = f2bf((q1 * c + q0 * s) * scale);
  kh[ob]     = f2bf(k0 * c - k1 * s);
  kh[ob + 1] = f2bf(k1 * c + k0 * s);
  vh[ob]     = f2bf(v0);
  vh[ob + 1] = f2bf(v1);
}

// ---------------------------------------------------------------------------
// 4) Flash attention: grid (N/128, B*H), block 256 (8 wave32s).
//    K tile copied global->LDS with CDNA5 async-to-LDS (ASYNCcnt), V tile
//    transposed via VALU while the async copy is in flight.
// ---------------------------------------------------------------------------
__global__ void __launch_bounds__(256)
attn_fwd_kernel(const __bf16* __restrict__ qh, const __bf16* __restrict__ kh,
                const __bf16* __restrict__ vh, __bf16* __restrict__ oh) {
  __shared__ __bf16 Kb[32 * 64];     // [key][dh] row-major
  __shared__ __bf16 Vt[64 * 32];     // transposed: [dh][key]
  __shared__ __bf16 Pl[8][16 * 32];  // per-wave P staging

  const int lane = threadIdx.x & 31;
  const int wave = threadIdx.x >> 5;
  const int half = lane >> 4;
  const int l16  = lane & 15;
  const int bh   = blockIdx.y;
  const int q0   = blockIdx.x * 128 + wave * 16;
  const size_t base = (size_t)bh * Nc * DHc;

  const unsigned kb_lds = (unsigned)(size_t)&Kb[threadIdx.x * 8];

  // Q fragments (scale prefolded)
  const __bf16* qrow = qh + base + (size_t)(q0 + l16) * DHc;
  v16bf qa0 = cat16(*(const v8bf*)(qrow + half * 16),
                    *(const v8bf*)(qrow + half * 16 + 8));
  v16bf qa1 = cat16(*(const v8bf*)(qrow + 32 + half * 16),
                    *(const v8bf*)(qrow + 32 + half * 16 + 8));

  v8f o0 = {}, o1 = {}, o2 = {}, o3 = {};
  float Mrun[8], Lrun[8];
#pragma unroll
  for (int j = 0; j < 8; ++j) { Mrun[j] = -1e30f; Lrun[j] = 0.0f; }

  for (int j0 = 0; j0 < Nc; j0 += 32) {
    // ---- K tile: async global->LDS; V tile: VALU transpose in its shadow ----
    {
      async_copy16(kb_lds, kh + base + (size_t)j0 * DHc + threadIdx.x * 8);
      const __bf16* vsrc = vh + base + (size_t)j0 * DHc;
      const int vr  = threadIdx.x >> 3;
      const int vc0 = (threadIdx.x & 7) * 8;
      v8bf vv = *(const v8bf*)(vsrc + vr * 64 + vc0);
#pragma unroll
      for (int e = 0; e < 8; ++e) Vt[(vc0 + e) * 32 + vr] = vv[e];
      wait_async0();
    }
    __syncthreads();

    // ---- S = Q * K^T ----
    v8f s0 = {}, s1 = {};
    {
      v16bf kb;
      kb = ld16c(&Kb[(l16) * 64 + half * 16]);           s0 = WMMA_BF16(qa0, kb, s0);
      kb = ld16c(&Kb[(l16) * 64 + 32 + half * 16]);      s0 = WMMA_BF16(qa1, kb, s0);
      kb = ld16c(&Kb[(16 + l16) * 64 + half * 16]);      s1 = WMMA_BF16(qa0, kb, s1);
      kb = ld16c(&Kb[(16 + l16) * 64 + 32 + half * 16]); s1 = WMMA_BF16(qa1, kb, s1);
    }

    // ---- online softmax ----
    float alpha[8];
#pragma unroll
    for (int jj = 0; jj < 8; ++jj) {
      const float mx = hmax16(fmaxf(s0[jj], s1[jj]));
      const float nm = fmaxf(Mrun[jj], mx);
      const float a  = __expf(Mrun[jj] - nm);
      const float p0 = __expf(s0[jj] - nm);
      const float p1 = __expf(s1[jj] - nm);
      const float rs = hsum16(p0 + p1);
      Lrun[jj]  = Lrun[jj] * a + rs;
      Mrun[jj]  = nm;
      alpha[jj] = a;
      s0[jj] = p0; s1[jj] = p1;
    }
#pragma unroll
    for (int jj = 0; jj < 8; ++jj) {
      o0[jj] *= alpha[jj]; o1[jj] *= alpha[jj];
      o2[jj] *= alpha[jj]; o3[jj] *= alpha[jj];
    }

    // ---- stage P (C-layout) -> LDS -> reload in A-layout ----
    __bf16* pw = &Pl[wave][0];
#pragma unroll
    for (int jj = 0; jj < 8; ++jj) {
      const int m = jj + 8 * half;
      pw[m * 32 + l16]      = f2bf(s0[jj]);
      pw[m * 32 + 16 + l16] = f2bf(s1[jj]);
    }
    const __bf16* prow = &pw[l16 * 32];
    v16bf pa = cat16(*(const v8bf*)(prow + half * 16),
                     *(const v8bf*)(prow + half * 16 + 8));

    // ---- O += P * V ----
    {
      v16bf vb;
      vb = ld16c(&Vt[(l16) * 32 + half * 16]);      o0 = WMMA_BF16(pa, vb, o0);
      vb = ld16c(&Vt[(16 + l16) * 32 + half * 16]); o1 = WMMA_BF16(pa, vb, o1);
      vb = ld16c(&Vt[(32 + l16) * 32 + half * 16]); o2 = WMMA_BF16(pa, vb, o2);
      vb = ld16c(&Vt[(48 + l16) * 32 + half * 16]); o3 = WMMA_BF16(pa, vb, o3);
    }
    __syncthreads();
  }

  // ---- normalize and store O as bf16 [B][N][H*DH] ----
  const int b = bh >> 4, h = bh & 15;
#pragma unroll
  for (int jj = 0; jj < 8; ++jj) {
    const float inv = 1.0f / Lrun[jj];
    const int m = q0 + jj + 8 * half;
    const size_t ob = ((size_t)b * Nc + m) * (Hc * DHc) + h * DHc;
    oh[ob + l16]      = f2bf(o0[jj] * inv);
    oh[ob + 16 + l16] = f2bf(o1[jj] * inv);
    oh[ob + 32 + l16] = f2bf(o2[jj] * inv);
    oh[ob + 48 + l16] = f2bf(o3[jj] * inv);
  }
}

// ---------------------------------------------------------------------------
// Launcher
// ---------------------------------------------------------------------------
extern "C" void kernel_launch(void* const* d_in, const int* in_sizes, int n_in,
                              void* d_out, int out_size, void* d_ws, size_t ws_size,
                              hipStream_t stream) {
  (void)in_sizes; (void)n_in; (void)out_size; (void)ws_size;
  const float* x     = (const float*)d_in[0];  // [B,N,DIM]
  const float* w_qkv = (const float*)d_in[1];  // [DIM, 3*H*DH]
  const float* w_out = (const float*)d_in[2];  // [H*DH, DIM]
  const float* b_out = (const float*)d_in[3];  // [DIM]
  float* out = (float*)d_out;                  // [B,N,DIM] f32

  const int M  = Bc * Nc;       // 4096
  const int K  = DIMc;          // 1024
  const int N3 = 3 * Hc * DHc;  // 3072
  const int HD = Hc * DHc;      // 1024

  char* ws = (char*)d_ws;
  auto take = [&](size_t bytes) {
    char* p = ws;
    ws += (bytes + 255) & ~(size_t)255;
    return p;
  };
  __bf16* xb  = (__bf16*)take((size_t)M * K * 2);
  __bf16* wqb = (__bf16*)take((size_t)K * N3 * 2);
  __bf16* wob = (__bf16*)take((size_t)HD * DIMc * 2);
  float*  qkv = (float*) take((size_t)M * N3 * 4);
  __bf16* qh  = (__bf16*)take((size_t)Bc * Hc * Nc * DHc * 2);
  __bf16* kh  = (__bf16*)take((size_t)Bc * Hc * Nc * DHc * 2);
  __bf16* vh  = (__bf16*)take((size_t)Bc * Hc * Nc * DHc * 2);
  __bf16* ohb = (__bf16*)take((size_t)M * HD * 2);

  // 1) casts
  {
    int n = M * K;
    cast_f32_bf16_kernel<<<(n + 255) / 256, 256, 0, stream>>>(x, xb, n);
    n = K * N3;
    cast_f32_bf16_kernel<<<(n + 255) / 256, 256, 0, stream>>>(w_qkv, wqb, n);
    n = HD * DIMc;
    cast_f32_bf16_kernel<<<(n + 255) / 256, 256, 0, stream>>>(w_out, wob, n);
  }
  // 2) QKV GEMM
  {
    dim3 grid(N3 / 128, M / 64);
    gemm_bf16_wmma<<<grid, 256, 0, stream>>>(xb, wqb, nullptr, qkv, M, K, N3);
  }
  // 3) RoPE + head split
  {
    const int total = Bc * Nc * Hc * (DHc / 2);
    rope_split_kernel<<<total / 256, 256, 0, stream>>>(qkv, qh, kh, vh);
  }
  // 4) attention
  {
    dim3 grid(Nc / 128, Bc * Hc);
    attn_fwd_kernel<<<grid, 256, 0, stream>>>(qh, kh, vh, ohb);
  }
  // 5) output projection + bias
  {
    dim3 grid(DIMc / 128, M / 64);
    gemm_bf16_wmma<<<grid, 256, 0, stream>>>(ohb, wob, b_out, out, M, HD, DIMc);
  }
}
